// DS_Block_67138928771417
// MI455X (gfx1250) — matrix operand order repaired and checked by reading, not compiled, not run on hardware
//
#include <hip/hip_runtime.h>
#include <hip/hip_bf16.h>
#include <math.h>

// ---------------------------------------------------------------------------
// DS_Block for MI455X (gfx1250, wave32, WMMA).
//  K1: qkv 1x1-conv GEMM (f16 WMMA, f32 acc) fused with 3-tap depthwise conv
//      along h (w==1), writes q/k/v as f16 [b][128][4096].
//  K2: per (b,head): S = Q Kt via f16 WMMA reading f16 rows directly from
//      global; fold 1/(||q|| ||k||) * temperature, 4x top-k masked softmax
//      combined into one 32x32 matrix, absorbed into proj:
//      M = proj_w * blkdiag(A).
//  K3: out = M @ V + proj_b  (128x128 @ 128x4096 f16 WMMA per batch, f32 out).
// B-fragment loads are batched/double-buffered so WMMA overlaps LDS/VMEM.
// ---------------------------------------------------------------------------

typedef __attribute__((ext_vector_type(16))) _Float16 v16h;
typedef __attribute__((ext_vector_type(8)))  _Float16 v8h;
typedef __attribute__((ext_vector_type(4)))  _Float16 v4h;
typedef __attribute__((ext_vector_type(8)))  float    v8f;

#define B_    32
#define DIM_  128
#define N_    4096
#define CH_   32     // channels per head

union AF { v16h v; v8h h[2]; };

__device__ __forceinline__ v8f wmma_f16(v16h a, v16h b, v8f c) {
  // D = A(16x32 f16) * B(32x16 f16) + C(16x16 f32)
  return __builtin_amdgcn_wmma_f32_16x16x32_f16(false, a, false, b,
                                                (short)0, c, false, false);
}

// ============================ K1: qkv + dwconv =============================
// grid (32 n-tiles, 6 o-blocks of 64, 32 batches), 128 threads (4 waves).
__launch_bounds__(128)
__global__ void k1_qkv_dwconv(const float* __restrict__ x,
                              const float* __restrict__ qkv_w,
                              const float* __restrict__ qkv_b,
                              const float* __restrict__ dw_w,
                              const float* __restrict__ dw_b,
                              _Float16* __restrict__ qh,
                              _Float16* __restrict__ kh,
                              _Float16* __restrict__ vh) {
  __shared__ alignas(16) _Float16 Asm[64][136];           // W rows, f16
  __shared__ alignas(16) union {
    _Float16 Xt[144][136];                                // x tile^T (n,c), f16
    float    Y[64][145];                                  // conv1x1 out, f32
  } su;

  const int t    = threadIdx.x;
  const int lane = t & 31;
  const int wv   = t >> 5;
  const int lo   = lane & 15;
  const int hi   = lane >> 4;
  const int n0   = blockIdx.x * 128;
  const int ob   = blockIdx.y * 64;
  const int b    = blockIdx.z;

  // ---- stage A: qkv_w rows [ob, ob+64), f32 -> f16, row-major
  {
    const int o = t >> 1, half = t & 1;
    const float4* src = (const float4*)(qkv_w + (size_t)(ob + o) * DIM_ + half * 64);
    #pragma unroll 4
    for (int i = 0; i < 16; ++i) {
      float4 v = src[i];
      v4h hv;
      hv[0] = (_Float16)v.x; hv[1] = (_Float16)v.y;
      hv[2] = (_Float16)v.z; hv[3] = (_Float16)v.w;
      *(v4h*)&Asm[o][half * 64 + i * 4] = hv;
    }
  }
  // ---- stage B: x tile with +/-8 halo, transposed to [n][c], zero padded
  {
    const int c = t;  // 0..127
    const float* xr = x + ((size_t)b * DIM_ + c) * N_;
    if (n0 >= 8 && n0 + 136 <= N_) {          // interior tile: no bounds checks
      #pragma unroll 4
      for (int j4 = 0; j4 < 144; j4 += 4) {
        float4 v = *(const float4*)(xr + n0 - 8 + j4);
        su.Xt[j4 + 0][c] = (_Float16)v.x; su.Xt[j4 + 1][c] = (_Float16)v.y;
        su.Xt[j4 + 2][c] = (_Float16)v.z; su.Xt[j4 + 3][c] = (_Float16)v.w;
      }
    } else {
      for (int j4 = 0; j4 < 144; j4 += 4) {
        const int n = n0 - 8 + j4;
        float v0, v1, v2, v3;
        v0 = (n     >= 0 && n     < N_) ? xr[n]     : 0.f;
        v1 = (n + 1 >= 0 && n + 1 < N_) ? xr[n + 1] : 0.f;
        v2 = (n + 2 >= 0 && n + 2 < N_) ? xr[n + 2] : 0.f;
        v3 = (n + 3 >= 0 && n + 3 < N_) ? xr[n + 3] : 0.f;
        su.Xt[j4 + 0][c] = (_Float16)v0; su.Xt[j4 + 1][c] = (_Float16)v1;
        su.Xt[j4 + 2][c] = (_Float16)v2; su.Xt[j4 + 3][c] = (_Float16)v3;
      }
    }
  }
  __syncthreads();

  // ---- GEMM: each wave does one 16-row m-tile x 9 n-tiles, K=128.
  // All 9 B-fragments of a k-step are preloaded so the in-order DScnt lets
  // each WMMA wait on a strictly smaller count (loads overlap matrix math).
  const int mb = wv * 16;
  v8f acc[9];
  #pragma unroll
  for (int j = 0; j < 9; ++j) acc[j] = (v8f){};
  #pragma unroll
  for (int ks = 0; ks < 4; ++ks) {
    const int kb = ks * 32;
    AF a;
    a.h[0] = *(const v8h*)&Asm[mb + lo][kb + hi * 16];
    a.h[1] = *(const v8h*)&Asm[mb + lo][kb + hi * 16 + 8];
    AF bf[9];
    #pragma unroll
    for (int j = 0; j < 9; ++j) {
      bf[j].h[0] = *(const v8h*)&su.Xt[j * 16 + lo][kb + hi * 16];
      bf[j].h[1] = *(const v8h*)&su.Xt[j * 16 + lo][kb + hi * 16 + 8];
    }
    #pragma unroll
    for (int j = 0; j < 9; ++j)
      acc[j] = wmma_f16(a.v, bf[j].v, acc[j]);
  }
  __syncthreads();   // done reading Xt; Y aliases it

  // ---- bias + spill to LDS (C-frag: M = r + 8*hi, N = 16*j + lo)
  float bias[8];
  #pragma unroll
  for (int r = 0; r < 8; ++r) bias[r] = qkv_b[ob + mb + r + 8 * hi];
  #pragma unroll
  for (int j = 0; j < 9; ++j)
    #pragma unroll
    for (int r = 0; r < 8; ++r)
      su.Y[mb + r + 8 * hi][j * 16 + lo] = acc[j][r] + bias[r];
  __syncthreads();

  // ---- 3-tap depthwise conv along n (kernel column kw=1), write f16
  {
    const int o  = t >> 1;            // local out channel 0..63
    const int og = ob + o;            // global 0..383
    const int seg = t & 1;            // n half: 0/1 (64 each)
    const float wl = dw_w[og * 9 + 1];
    const float wc = dw_w[og * 9 + 4];
    const float wr = dw_w[og * 9 + 7];
    const float bb = dw_b[og];
    const int oc = og & 127;
    _Float16* dst = (og < 128) ? qh : (og < 256) ? kh : vh;
    dst += ((size_t)b * DIM_ + oc) * N_;
    const float* Yp = &su.Y[o][0];
    for (int i8 = 0; i8 < 64; i8 += 8) {
      v8h outv;
      #pragma unroll
      for (int e = 0; e < 8; ++e) {
        const int n  = n0 + seg * 64 + i8 + e;
        const int jj = 8 + seg * 64 + i8 + e;
        const float l_ = (n >= 1)      ? Yp[jj - 1] : 0.f;
        const float c_ = Yp[jj];
        const float r_ = (n <= N_ - 2) ? Yp[jj + 1] : 0.f;
        outv[e] = (_Float16)(wl * l_ + wc * c_ + wr * r_ + bb);
      }
      *(v8h*)(dst + n0 + seg * 64 + i8) = outv;
    }
  }
}

// ==================== K2: attention + topk softmax + fold ==================
__device__ __forceinline__ void ld_qk(const _Float16* __restrict__ Q,
                                      const _Float16* __restrict__ K,
                                      int lo, int hi, int kb,
                                      AF& a0, AF& a1, AF& b0, AF& b1) {
  a0.h[0] = *(const v8h*)(Q + (size_t)(lo)      * N_ + kb + hi * 16);
  a0.h[1] = *(const v8h*)(Q + (size_t)(lo)      * N_ + kb + hi * 16 + 8);
  a1.h[0] = *(const v8h*)(Q + (size_t)(16 + lo) * N_ + kb + hi * 16);
  a1.h[1] = *(const v8h*)(Q + (size_t)(16 + lo) * N_ + kb + hi * 16 + 8);
  b0.h[0] = *(const v8h*)(K + (size_t)(lo)      * N_ + kb + hi * 16);
  b0.h[1] = *(const v8h*)(K + (size_t)(lo)      * N_ + kb + hi * 16 + 8);
  b1.h[0] = *(const v8h*)(K + (size_t)(16 + lo) * N_ + kb + hi * 16);
  b1.h[1] = *(const v8h*)(K + (size_t)(16 + lo) * N_ + kb + hi * 16 + 8);
}

// grid (4 heads, 32 batches), 32 threads (1 wave).
__launch_bounds__(32)
__global__ void k2_attn(const _Float16* __restrict__ qh,
                        const _Float16* __restrict__ kh,
                        const float* __restrict__ proj_w,
                        const float* __restrict__ temperature,
                        const float* __restrict__ attn_w,
                        _Float16* __restrict__ Mh) {
  __shared__ float Ssm[32][33];
  __shared__ float Afin[32][33];
  __shared__ float invn[64];          // [0..31]=1/||q||, [32..63]=1/||k||
  const int lane = threadIdx.x;
  const int lo = lane & 15, hi = lane >> 4;
  const int hd = blockIdx.x, b = blockIdx.y;
  const _Float16* Q = qh + ((size_t)b * DIM_ + hd * CH_) * N_;
  const _Float16* K = kh + ((size_t)b * DIM_ + hd * CH_) * N_;

  // ---- row norms (computed on exactly the f16 data the GEMM consumes)
  {
    const _Float16* rq = Q + (size_t)lane * N_;
    const _Float16* rk = K + (size_t)lane * N_;
    float sq = 0.f, sk = 0.f;
    for (int i = 0; i < N_ / 8; ++i) {
      v8h a = *(const v8h*)(rq + i * 8);
      v8h c = *(const v8h*)(rk + i * 8);
      #pragma unroll
      for (int e = 0; e < 8; ++e) {
        float fa = (float)a[e]; sq += fa * fa;
        float fc = (float)c[e]; sk += fc * fc;
      }
    }
    invn[lane]      = 1.f / fmaxf(sqrtf(sq), 1e-12f);
    invn[32 + lane] = 1.f / fmaxf(sqrtf(sk), 1e-12f);
  }
  __syncthreads();

  // ---- S = Q * K^T, 2-deep software pipeline on the fragment sets
  v8f acc[2][2];
  acc[0][0] = (v8f){}; acc[0][1] = (v8f){}; acc[1][0] = (v8f){}; acc[1][1] = (v8f){};
  AF a0A, a1A, b0A, b1A, a0B, a1B, b0B, b1B;
  ld_qk(Q, K, lo, hi, 0, a0A, a1A, b0A, b1A);
  for (int ks = 0; ks < N_ / 32; ks += 2) {
    ld_qk(Q, K, lo, hi, (ks + 1) * 32, a0B, a1B, b0B, b1B);
    acc[0][0] = wmma_f16(a0A.v, b0A.v, acc[0][0]);
    acc[0][1] = wmma_f16(a0A.v, b1A.v, acc[0][1]);
    acc[1][0] = wmma_f16(a1A.v, b0A.v, acc[1][0]);
    acc[1][1] = wmma_f16(a1A.v, b1A.v, acc[1][1]);
    const int kn = (ks + 2 < N_ / 32) ? (ks + 2) * 32 : 0;  // tail reload: unused
    ld_qk(Q, K, lo, hi, kn, a0A, a1A, b0A, b1A);
    acc[0][0] = wmma_f16(a0B.v, b0B.v, acc[0][0]);
    acc[0][1] = wmma_f16(a0B.v, b1B.v, acc[0][1]);
    acc[1][0] = wmma_f16(a1B.v, b0B.v, acc[1][0]);
    acc[1][1] = wmma_f16(a1B.v, b1B.v, acc[1][1]);
  }
  const float temp = temperature[hd];
  #pragma unroll
  for (int mi = 0; mi < 2; ++mi)
    #pragma unroll
    for (int ni = 0; ni < 2; ++ni)
      #pragma unroll
      for (int r = 0; r < 8; ++r) {
        const int row = 16 * mi + r + 8 * hi;
        const int col = 16 * ni + lo;
        Ssm[row][col] = acc[mi][ni][r] * invn[row] * invn[32 + col] * temp;
      }
  __syncthreads();

  // ---- combined top-k (16,21,24,25) masked softmax, weights attn_w
  const float w0 = attn_w[0], w1 = attn_w[1], w2 = attn_w[2], w3 = attn_w[3];
  for (int row = 0; row < 32; ++row) {
    const float s = Ssm[row][lane];
    int rank = 0;
    #pragma unroll 8
    for (int i = 0; i < 32; ++i) {
      const float si = Ssm[row][i];                  // LDS broadcast
      rank += (si > s) || (si == s && i < lane);     // stable top-k
    }
    float m = s;
    #pragma unroll
    for (int off = 16; off; off >>= 1) m = fmaxf(m, __shfl_xor(m, off, 32));
    const float e = expf(s - m);
    float d0 = (rank < 16) ? e : 0.f, d1 = (rank < 21) ? e : 0.f;
    float d2 = (rank < 24) ? e : 0.f, d3 = (rank < 25) ? e : 0.f;
    #pragma unroll
    for (int off = 16; off; off >>= 1) {
      d0 += __shfl_xor(d0, off, 32); d1 += __shfl_xor(d1, off, 32);
      d2 += __shfl_xor(d2, off, 32); d3 += __shfl_xor(d3, off, 32);
    }
    const float a = e * (((rank < 16) ? w0 / d0 : 0.f) +
                         ((rank < 21) ? w1 / d1 : 0.f) +
                         ((rank < 24) ? w2 / d2 : 0.f) +
                         ((rank < 25) ? w3 / d3 : 0.f));
    Afin[row][lane] = a;
  }
  __syncthreads();

  // ---- fold into projection: M[o][hd*32+d] = sum_j proj_w[o][hd*32+j]*A[j][d]
  for (int o = 0; o < DIM_; ++o) {
    const float* pr = proj_w + (size_t)o * DIM_ + hd * CH_;   // uniform -> s_load
    float av = 0.f;
    #pragma unroll
    for (int j = 0; j < 32; ++j) av += pr[j] * Afin[j][lane];
    Mh[((size_t)b * DIM_ + o) * DIM_ + hd * CH_ + lane] = (_Float16)av;
  }
}

// ======================= K3: out = M @ V + proj_b ==========================
// grid (32 n-tiles of 128, 32 batches), 256 threads (8 waves).
__launch_bounds__(256)
__global__ void k3_proj(const _Float16* __restrict__ Mh,
                        const _Float16* __restrict__ vh,
                        const float* __restrict__ proj_b,
                        float* __restrict__ out) {
  __shared__ alignas(16) _Float16 Vt[128][136];     // V tile^T (n,c)
  const int t = threadIdx.x, lane = t & 31, wv = t >> 5;
  const int lo = lane & 15, hi = lane >> 4;
  const int n0 = blockIdx.x * 128, b = blockIdx.y;

  // ---- stage V tile transposed into LDS
  {
    const int c = t & 127, seg = t >> 7;            // seg: n half of 64
    const _Float16* src = vh + ((size_t)b * DIM_ + c) * N_ + n0 + seg * 64;
    #pragma unroll
    for (int i = 0; i < 8; ++i) {
      v8h v = *(const v8h*)(src + i * 8);
      const int nn = seg * 64 + i * 8;
      #pragma unroll
      for (int e = 0; e < 8; ++e) Vt[nn + e][c] = v[e];
    }
  }
  __syncthreads();

  // ---- each wave: 16-row m-tile x 8 n-tiles, K = 128 (batched B preload)
  const int mb = wv * 16;
  const _Float16* Arow = Mh + ((size_t)b * DIM_ + mb + lo) * DIM_;
  v8f acc[8];
  #pragma unroll
  for (int j = 0; j < 8; ++j) acc[j] = (v8f){};
  #pragma unroll
  for (int ks = 0; ks < 4; ++ks) {
    const int kb = ks * 32;
    AF a;
    a.h[0] = *(const v8h*)(Arow + kb + hi * 16);
    a.h[1] = *(const v8h*)(Arow + kb + hi * 16 + 8);
    AF bf[8];
    #pragma unroll
    for (int j = 0; j < 8; ++j) {
      bf[j].h[0] = *(const v8h*)&Vt[j * 16 + lo][kb + hi * 16];
      bf[j].h[1] = *(const v8h*)&Vt[j * 16 + lo][kb + hi * 16 + 8];
    }
    #pragma unroll
    for (int j = 0; j < 8; ++j)
      acc[j] = wmma_f16(a.v, bf[j].v, acc[j]);
  }

  float pb[8];
  #pragma unroll
  for (int r = 0; r < 8; ++r) pb[r] = proj_b[mb + r + 8 * hi];
  #pragma unroll
  for (int j = 0; j < 8; ++j)
    #pragma unroll
    for (int r = 0; r < 8; ++r)
      out[((size_t)b * DIM_ + mb + r + 8 * hi) * N_ + n0 + j * 16 + lo] =
          acc[j][r] + pb[r];
}

// ================================ launch ===================================
extern "C" void kernel_launch(void* const* d_in, const int* in_sizes, int n_in,
                              void* d_out, int out_size, void* d_ws, size_t ws_size,
                              hipStream_t stream) {
  (void)in_sizes; (void)n_in; (void)out_size; (void)ws_size;
  const float* x      = (const float*)d_in[0];
  const float* qkv_w  = (const float*)d_in[1];
  const float* qkv_b  = (const float*)d_in[2];
  const float* dw_w   = (const float*)d_in[3];
  const float* dw_b   = (const float*)d_in[4];
  const float* proj_w = (const float*)d_in[5];
  const float* proj_b = (const float*)d_in[6];
  const float* temp   = (const float*)d_in[7];
  const float* attn_w = (const float*)d_in[8];

  const size_t qkvBytes = (size_t)B_ * DIM_ * N_ * sizeof(_Float16);  // 32 MB
  char* ws = (char*)d_ws;
  _Float16* qh = (_Float16*)(ws);
  _Float16* kh = (_Float16*)(ws + qkvBytes);
  _Float16* vh = (_Float16*)(ws + 2 * qkvBytes);
  _Float16* Mh = (_Float16*)(ws + 3 * qkvBytes);                      // 1 MB

  dim3 g1(N_ / 128, 6, B_);
  k1_qkv_dwconv<<<g1, 128, 0, stream>>>(x, qkv_w, qkv_b, dw_w, dw_b, qh, kh, vh);

  dim3 g2(4, B_);
  k2_attn<<<g2, 32, 0, stream>>>(qh, kh, proj_w, temp, attn_w, Mh);

  dim3 g3(N_ / 128, B_);
  k3_proj<<<g3, 256, 0, stream>>>(Mh, vh, proj_b, (float*)d_out);
}
